// DGCNN3_33105607917643
// MI455X (gfx1250) — compile-verified, matrix operand order checked by reading, not run on hardware
//
#include <hip/hip_runtime.h>
#include <hip/hip_bf16.h>

// ---------------------------------------------------------------------------
// DGCNN inference for MI455X (gfx1250, wave32, WMMA 16x16x32 f16 -> f32 acc)
// ---------------------------------------------------------------------------

#define B_GR   64
#define NPG    1024
#define NTOT   (B_GR * NPG)   // 65536 nodes
#define KNN    5
#define EPSBN  1e-5f

#ifndef __has_builtin
#define __has_builtin(x) 0
#endif
#if __has_builtin(__builtin_amdgcn_tensor_load_to_lds)
#define HAVE_TDM 1
#else
#define HAVE_TDM 0
#endif

typedef __attribute__((ext_vector_type(16))) _Float16 v16h;
typedef __attribute__((ext_vector_type(8)))  _Float16 v8h;
typedef __attribute__((ext_vector_type(8)))  float    v8f;
typedef __attribute__((ext_vector_type(4)))  unsigned int u32x4;
typedef __attribute__((ext_vector_type(8)))  int          i32x8;
typedef __attribute__((ext_vector_type(4)))  int          i32x4;

// ---------------------------------------------------------------------------
// Fragment loader: 16x32 f16 A-fragment (or 32x16 B-fragment from a row-major
// (N,K) weight array) per CDNA5 ISA layout:
//   lane l holds row/col (l&15); lanes<16: K[8h..8h+7] in v0-3, K[16+8h..] v4-7.
// Works for both LDS-staged tiles and global f16 arrays (address space is
// recovered after inlining -> ds_load_b128 / global_load_b128).
// ---------------------------------------------------------------------------
__device__ __forceinline__ v16h frag_ld(const _Float16* p, int strideH,
                                        int row0, int k0, int lane) {
  int r = lane & 15;
  int h = (lane >> 4) & 1;
  const _Float16* rp = p + (size_t)(row0 + r) * strideH + k0 + h * 8;
  v8h lo = *(const v8h*)(rp);        // K [k0+8h .. +7]
  v8h hi = *(const v8h*)(rp + 16);   // K [k0+16+8h .. +7]
  v16h out;
#pragma unroll
  for (int i = 0; i < 8; ++i) { out[i] = lo[i]; out[i + 8] = hi[i]; }
  return out;
}

#define WMMA_F16(A, Bf, C) \
  __builtin_amdgcn_wmma_f32_16x16x32_f16(false, (A), false, (Bf), (short)0, (C), false, false)

// ---------------------------------------------------------------------------
// Prep: fold BatchNorm (eval) into weights, convert to f16 with K padding.
//   scale = g * rsqrt(var+eps); W' = scale*W (row); b' = scale*b + be - mu*scale
// g==nullptr -> plain linear (scale=1, off=0).
// ---------------------------------------------------------------------------
__global__ void k_fold_f16(const float* __restrict__ W, const float* __restrict__ b,
                           const float* __restrict__ g, const float* __restrict__ be,
                           const float* __restrict__ mu, const float* __restrict__ var,
                           int dout, int din, int ldp,
                           _Float16* __restrict__ Wout, float* __restrict__ bout) {
  int total = dout * ldp;
  for (int i = blockIdx.x * blockDim.x + threadIdx.x; i < total;
       i += gridDim.x * blockDim.x) {
    int r = i / ldp, c = i - r * ldp;
    float s = 1.0f, off = 0.0f;
    if (g) { s = g[r] * rsqrtf(var[r] + EPSBN); off = be[r] - mu[r] * s; }
    Wout[i] = (_Float16)(c < din ? W[(size_t)r * din + c] * s : 0.0f);
    if (c == 0) bout[r] = b[r] * s + off;
  }
}

__global__ void k_fold_f32(const float* __restrict__ W, const float* __restrict__ b,
                           const float* __restrict__ g, const float* __restrict__ be,
                           const float* __restrict__ mu, const float* __restrict__ var,
                           int dout, int din,
                           float* __restrict__ Wout, float* __restrict__ bout) {
  int total = dout * din;
  for (int i = blockIdx.x * blockDim.x + threadIdx.x; i < total;
       i += gridDim.x * blockDim.x) {
    int r = i / din;
    float s = 1.0f, off = 0.0f;
    if (g) { s = g[r] * rsqrtf(var[r] + EPSBN); off = be[r] - mu[r] * s; }
    Wout[i] = W[i] * s;
    if (i - r * din == 0) bout[r] = b[r] * s + off;
  }
}

// ---------------------------------------------------------------------------
// Pack xx = concat(x, pos) into f16 rows padded K 4->32, plus f32 sq-norms.
// ---------------------------------------------------------------------------
__global__ void k_pack_xx(const float* __restrict__ x, const float* __restrict__ pos,
                          _Float16* __restrict__ xxh, float* __restrict__ d2) {
  for (int i = blockIdx.x * blockDim.x + threadIdx.x; i < NTOT;
       i += gridDim.x * blockDim.x) {
    float f0 = x[i], f1 = pos[i * 3], f2 = pos[i * 3 + 1], f3 = pos[i * 3 + 2];
    _Float16* r = xxh + (size_t)i * 32;
    r[0] = (_Float16)f0; r[1] = (_Float16)f1; r[2] = (_Float16)f2; r[3] = (_Float16)f3;
#pragma unroll
    for (int c = 4; c < 32; ++c) r[c] = (_Float16)0.0f;
    d2[i] = f0 * f0 + f1 * f1 + f2 * f2 + f3 * f3;
  }
}

__global__ void k_rownorm64(const float* __restrict__ x1, float* __restrict__ d2) {
  for (int i = blockIdx.x * blockDim.x + threadIdx.x; i < NTOT;
       i += gridDim.x * blockDim.x) {
    const float* r = x1 + (size_t)i * 64;
    float s = 0.0f;
#pragma unroll 8
    for (int c = 0; c < 64; ++c) s += r[c] * r[c];
    d2[i] = s;
  }
}

// ---------------------------------------------------------------------------
// kNN: one wave per (graph, 16-row strip). S = F F^T via WMMA, dist strip to
// LDS, per-half-row top-5 insertion (float4 scan), 2-way merge.
// KSTEPS compile-time: 1 (d padded to 32) or 2 (d = 64) -> branch-free loop.
// ---------------------------------------------------------------------------
template <int KSTEPS>
__global__ __launch_bounds__(32) void k_knn(const _Float16* __restrict__ feat,
                                            int strideH, const float* __restrict__ d2,
                                            int* __restrict__ idx) {
  __shared__ __align__(16) float distS[16 * NPG];       // 64 KB strip
  __shared__ float mv[32 * KNN];
  __shared__ int   mi[32 * KNN];
  int lane = threadIdx.x;
  int g = blockIdx.x >> 6, rt = blockIdx.x & 63;
  int rowBase = g * NPG + rt * 16;
  int h = (lane >> 4) & 1;

  v16h a[KSTEPS];
#pragma unroll
  for (int kb = 0; kb < KSTEPS; ++kb)
    a[kb] = frag_ld(feat, strideH, rowBase, kb * 32, lane);

  float d2r[8];
#pragma unroll
  for (int r = 0; r < 8; ++r) d2r[r] = d2[rowBase + h * 8 + r];

  for (int ct = 0; ct < 64; ++ct) {
    int colBase = g * NPG + ct * 16;
    v8f c = {};
#pragma unroll
    for (int kb = 0; kb < KSTEPS; ++kb) {
      v16h b = frag_ld(feat, strideH, colBase, kb * 32, lane);
      c = WMMA_F16(a[kb], b, c);
    }
    float d2c = d2[colBase + (lane & 15)];
#pragma unroll
    for (int r = 0; r < 8; ++r) {
      float dist = d2r[r] + d2c - 2.0f * c[r];
      distS[(r + 8 * h) * NPG + ct * 16 + (lane & 15)] = dist;
    }
  }
  __syncthreads();

  int row = lane & 15;
  float tv[KNN]; int ti[KNN];
#pragma unroll
  for (int q = 0; q < KNN; ++q) { tv[q] = 3.0e38f; ti[q] = 0; }
  const float4* rp4 = (const float4*)(distS + row * NPG + h * 512);
  for (int j4 = 0; j4 < 128; ++j4) {
    float4 v4 = rp4[j4];
    float vv[4] = {v4.x, v4.y, v4.z, v4.w};
#pragma unroll
    for (int u = 0; u < 4; ++u) {
      float v = vv[u];
      if (v < tv[KNN - 1]) {
        int p = KNN - 1;
        while (p > 0 && v < tv[p - 1]) { tv[p] = tv[p - 1]; ti[p] = ti[p - 1]; --p; }
        tv[p] = v; ti[p] = h * 512 + j4 * 4 + u;
      }
    }
  }
#pragma unroll
  for (int q = 0; q < KNN; ++q) { mv[lane * KNN + q] = tv[q]; mi[lane * KNN + q] = ti[q]; }
  __syncthreads();

  if (lane < 16) {
    const float* va = &mv[lane * KNN];        const int* ia = &mi[lane * KNN];
    const float* vb = &mv[(lane + 16) * KNN]; const int* ib = &mi[(lane + 16) * KNN];
    int pa = 0, pb = 0;
    int node = rowBase + lane;
#pragma unroll
    for (int q = 0; q < KNN; ++q) {
      bool ta = (pb >= KNN) || (pa < KNN && va[pa] <= vb[pb]);
      int sel = ta ? ia[pa] : ib[pb];
      if (ta) ++pa; else ++pb;
      idx[node * KNN + q] = g * NPG + sel;
    }
  }
}

// ---------------------------------------------------------------------------
// conv1 EdgeConv: one wave per 16 nodes (= 80 edges = 5 exact M-tiles).
// MLP 8(->32 pad) -> 64 -> 64 -> 64 via WMMA, BN folded, sum over k=5.
// Writes x1 f32 and f16 into feat[.,0:64) (row stride 192).
// ---------------------------------------------------------------------------
__global__ __launch_bounds__(32) void k_conv1(
    const float* __restrict__ x, const float* __restrict__ pos,
    const int* __restrict__ idx,
    const _Float16* __restrict__ W0, const float* __restrict__ b0,
    const _Float16* __restrict__ W1, const float* __restrict__ b1,
    const _Float16* __restrict__ W2, const float* __restrict__ b2,
    float* __restrict__ x1, _Float16* __restrict__ feat) {
  __shared__ __align__(16) _Float16 Ain[80 * 32];
  __shared__ __align__(16) _Float16 ActA[80 * 64];
  __shared__ __align__(16) _Float16 ActB[80 * 64];
  __shared__ __align__(16) float    Out[80 * 64];
  int lane = threadIdx.x;
  int nodeBase = blockIdx.x * 16;
  int rr = lane & 15, hh = (lane >> 4) & 1;

  for (int e = lane; e < 80; e += 32) {
    int node = nodeBase + e / KNN;
    int j = idx[node * KNN + (e % KNN)];
    float fi0 = x[node], fi1 = pos[node * 3], fi2 = pos[node * 3 + 1], fi3 = pos[node * 3 + 2];
    float fj0 = x[j],    fj1 = pos[j * 3],    fj2 = pos[j * 3 + 1],    fj3 = pos[j * 3 + 2];
    _Float16* r = Ain + e * 32;
    r[0] = (_Float16)fi0; r[1] = (_Float16)fi1; r[2] = (_Float16)fi2; r[3] = (_Float16)fi3;
    r[4] = (_Float16)(fj0 - fi0); r[5] = (_Float16)(fj1 - fi1);
    r[6] = (_Float16)(fj2 - fi2); r[7] = (_Float16)(fj3 - fi3);
#pragma unroll
    for (int c = 8; c < 32; ++c) r[c] = (_Float16)0.0f;
  }
  __syncthreads();

  // ----- Layer 0: K=32 (padded from 8), N=64, BN+ReLU -----
  {
    v16h bfr[4];
#pragma unroll
    for (int nt = 0; nt < 4; ++nt) bfr[nt] = frag_ld(W0, 32, nt * 16, 0, lane);
    for (int mt = 0; mt < 5; ++mt) {
      v16h a = frag_ld(Ain, 32, mt * 16, 0, lane);
#pragma unroll
      for (int nt = 0; nt < 4; ++nt) {
        v8f c = {};
        c = WMMA_F16(a, bfr[nt], c);
        float bias = b0[nt * 16 + rr];
#pragma unroll
        for (int r = 0; r < 8; ++r) {
          float v = c[r] + bias; v = v > 0.0f ? v : 0.0f;
          ActA[(mt * 16 + r + 8 * hh) * 64 + nt * 16 + rr] = (_Float16)v;
        }
      }
    }
  }
  __syncthreads();

  // ----- Layer 1: K=64, N=64, BN+ReLU -----
  {
    v16h bfr[4][2];
#pragma unroll
    for (int nt = 0; nt < 4; ++nt) {
      bfr[nt][0] = frag_ld(W1, 64, nt * 16, 0, lane);
      bfr[nt][1] = frag_ld(W1, 64, nt * 16, 32, lane);
    }
    for (int mt = 0; mt < 5; ++mt) {
      v16h a0 = frag_ld(ActA, 64, mt * 16, 0, lane);
      v16h a1 = frag_ld(ActA, 64, mt * 16, 32, lane);
#pragma unroll
      for (int nt = 0; nt < 4; ++nt) {
        v8f c = {};
        c = WMMA_F16(a0, bfr[nt][0], c);
        c = WMMA_F16(a1, bfr[nt][1], c);
        float bias = b1[nt * 16 + rr];
#pragma unroll
        for (int r = 0; r < 8; ++r) {
          float v = c[r] + bias; v = v > 0.0f ? v : 0.0f;
          ActB[(mt * 16 + r + 8 * hh) * 64 + nt * 16 + rr] = (_Float16)v;
        }
      }
    }
  }
  __syncthreads();

  // ----- Layer 2: K=64, N=64, plain linear -> f32 Out -----
  {
    v16h bfr[4][2];
#pragma unroll
    for (int nt = 0; nt < 4; ++nt) {
      bfr[nt][0] = frag_ld(W2, 64, nt * 16, 0, lane);
      bfr[nt][1] = frag_ld(W2, 64, nt * 16, 32, lane);
    }
    for (int mt = 0; mt < 5; ++mt) {
      v16h a0 = frag_ld(ActB, 64, mt * 16, 0, lane);
      v16h a1 = frag_ld(ActB, 64, mt * 16, 32, lane);
#pragma unroll
      for (int nt = 0; nt < 4; ++nt) {
        v8f c = {};
        c = WMMA_F16(a0, bfr[nt][0], c);
        c = WMMA_F16(a1, bfr[nt][1], c);
        float bias = b2[nt * 16 + rr];
#pragma unroll
        for (int r = 0; r < 8; ++r)
          Out[(mt * 16 + r + 8 * hh) * 64 + nt * 16 + rr] = c[r] + bias;
      }
    }
  }
  __syncthreads();

  // ----- add-aggregation over k=5 -----
  for (int t = lane; t < 16 * 64; t += 32) {
    int nl = t >> 6, ch = t & 63;
    float s = 0.0f;
#pragma unroll
    for (int k2 = 0; k2 < KNN; ++k2) s += Out[(nl * KNN + k2) * 64 + ch];
    int node = nodeBase + nl;
    x1[(size_t)node * 64 + ch] = s;
    feat[(size_t)node * 192 + ch] = (_Float16)s;
  }
}

// ---------------------------------------------------------------------------
// conv2 EdgeConv: 128 -> 128 plain linear, sum over k=5.
// Weights (128x128 f16 = 32 KB) staged LDS-resident via the Tensor Data Mover
// (one 2-D descriptor, tile == tensor, data_size = 2 B, TENSORcnt-tracked),
// overlapping the edge-feature gather. Falls back to direct global B-fragment
// loads when the TDM builtin is unavailable.
// ---------------------------------------------------------------------------
__global__ __launch_bounds__(32) void k_conv2(const float* __restrict__ x1,
                                              const int* __restrict__ idx,
                                              const _Float16* __restrict__ W,
                                              const float* __restrict__ b,
                                              _Float16* __restrict__ feat) {
  __shared__ __align__(16) _Float16 Ain[80 * 128];   // 20 KB
  __shared__ __align__(16) float    Out[80 * 128];   // 40 KB
#if HAVE_TDM
  __shared__ __align__(16) _Float16 Ws[128 * 128];   // 32 KB TDM destination
#endif
  int lane = threadIdx.x;
  int nodeBase = blockIdx.x * 16;
  int rr = lane & 15, hh = (lane >> 4) & 1;
  const _Float16* Wsrc = W;

#if HAVE_TDM
  {
    unsigned long long ga = (unsigned long long)(uintptr_t)W;
    unsigned ldsa = (unsigned)(uintptr_t)&Ws[0];     // LDS byte offset (addr[31:0])
    u32x4 g0;
    g0[0] = 1u;                                       // count=1, user descriptor
    g0[1] = ldsa;                                     // lds_addr
    g0[2] = (unsigned)(ga & 0xFFFFFFFFu);             // global_addr[31:0]
    g0[3] = (unsigned)((ga >> 32) & 0x01FFFFFFu) | (2u << 30);  // addr hi | type=2
    i32x8 g1;
    g1[0] = (int)(1u << 16);                          // wg_mask=0, data_size=1 (2 B)
    g1[1] = (int)(128u << 16);                        // tensor_dim0 = 128
    g1[2] = (int)(128u << 16);                        // tensor_dim1 = 128
    g1[3] = (int)(128u << 16);                        // tile_dim0 = 128
    g1[4] = (int)128u;                                // tile_dim1 = 128, tile_dim2 = 0
    g1[5] = (int)128u;                                // tensor_dim0_stride = 128
    g1[6] = 0;
    g1[7] = 0;
    i32x4 z4 = {0, 0, 0, 0};
#if __clang_major__ >= 23
    i32x8 z8 = {0, 0, 0, 0, 0, 0, 0, 0};
    __builtin_amdgcn_tensor_load_to_lds(g0, g1, z4, z4, z8, 0);
#else
    __builtin_amdgcn_tensor_load_to_lds(g0, g1, z4, z4, 0);
#endif
  }
#endif

  for (int t = lane; t < 80 * 64; t += 32) {
    int e = t >> 6, c = t & 63;
    int node = nodeBase + e / KNN;
    int j = idx[node * KNN + (e % KNN)];
    float xi = x1[(size_t)node * 64 + c];
    float xj = x1[(size_t)j * 64 + c];
    Ain[e * 128 + c]      = (_Float16)xi;
    Ain[e * 128 + 64 + c] = (_Float16)(xj - xi);
  }

#if HAVE_TDM
#if __has_builtin(__builtin_amdgcn_s_wait_tensorcnt)
  __builtin_amdgcn_s_wait_tensorcnt(0);
#endif
  Wsrc = Ws;
#endif
  __syncthreads();

  for (int nt = 0; nt < 8; ++nt) {
    v16h bfr[4];
#pragma unroll
    for (int kb = 0; kb < 4; ++kb) bfr[kb] = frag_ld(Wsrc, 128, nt * 16, kb * 32, lane);
    float bias = b[nt * 16 + rr];
    for (int mt = 0; mt < 5; ++mt) {
      v8f c = {};
#pragma unroll
      for (int kb = 0; kb < 4; ++kb) {
        v16h a = frag_ld(Ain, 128, mt * 16, kb * 32, lane);
        c = WMMA_F16(a, bfr[kb], c);
      }
#pragma unroll
      for (int r = 0; r < 8; ++r)
        Out[(mt * 16 + r + 8 * hh) * 128 + nt * 16 + rr] = c[r] + bias;
    }
  }
  __syncthreads();

  for (int t = lane; t < 16 * 128; t += 32) {
    int nl = t >> 7, ch = t & 127;
    float s = 0.0f;
#pragma unroll
    for (int k2 = 0; k2 < KNN; ++k2) s += Out[(nl * KNN + k2) * 128 + ch];
    feat[(size_t)(nodeBase + nl) * 192 + 64 + ch] = (_Float16)s;
  }
}

// ---------------------------------------------------------------------------
// lin1 (192->1024) fused with global max pool. One wave per (graph, 16 outs).
// Weight-stationary: 6 K-block B fragments held in VGPRs for the whole graph.
// max_n(h + b) = max_n(h) + b  (bias per column), so bias added after pooling.
// ---------------------------------------------------------------------------
__global__ __launch_bounds__(32) void k_lin1max(const _Float16* __restrict__ feat,
                                                const _Float16* __restrict__ W,
                                                const float* __restrict__ bias,
                                                float* __restrict__ pooled) {
  __shared__ float sm[32];
  int lane = threadIdx.x;
  int g = blockIdx.x >> 6, nt = blockIdx.x & 63;

  v16h bfr[6];
#pragma unroll
  for (int kb = 0; kb < 6; ++kb) bfr[kb] = frag_ld(W, 192, nt * 16, kb * 32, lane);

  v8f mx;
#pragma unroll
  for (int r = 0; r < 8; ++r) mx[r] = -3.0e38f;

  for (int mt = 0; mt < 64; ++mt) {
    int rowBase = g * NPG + mt * 16;
    if (mt + 1 < 64)  // warm next A strip into cache (global_prefetch_b8)
      __builtin_prefetch(feat + (size_t)(rowBase + 16 + lane) * 192, 0, 1);
    v8f c = {};
#pragma unroll
    for (int kb = 0; kb < 6; ++kb) {
      v16h a = frag_ld(feat, 192, rowBase, kb * 32, lane);
      c = WMMA_F16(a, bfr[kb], c);
    }
#pragma unroll
    for (int r = 0; r < 8; ++r) mx[r] = fmaxf(mx[r], c[r]);
  }

  float m = mx[0];
#pragma unroll
  for (int r = 1; r < 8; ++r) m = fmaxf(m, mx[r]);
  sm[lane] = m;
  __syncthreads();
  if (lane < 16) {
    float v = fmaxf(sm[lane], sm[lane + 16]) + bias[nt * 16 + lane];
    pooled[g * NPG + nt * 16 + lane] = v;
  }
}

// ---------------------------------------------------------------------------
// Head MLP: 1024 -> 512 (BN+ReLU) -> 256 (BN+ReLU) -> 8, f32 exact.
// ---------------------------------------------------------------------------
__global__ __launch_bounds__(256) void k_mlp_head(
    const float* __restrict__ pooled,
    const float* __restrict__ W0, const float* __restrict__ b0,
    const float* __restrict__ W1, const float* __restrict__ b1,
    const float* __restrict__ W2, const float* __restrict__ b2,
    float* __restrict__ out) {
  __shared__ float h0[1024];
  __shared__ float h1[512];
  __shared__ float h2[256];
  int g = blockIdx.x, t = threadIdx.x;
  for (int i = t; i < 1024; i += 256) h0[i] = pooled[g * 1024 + i];
  __syncthreads();
  for (int o = t; o < 512; o += 256) {
    float s = b0[o];
    const float* w = W0 + (size_t)o * 1024;
#pragma unroll 8
    for (int c = 0; c < 1024; ++c) s += h0[c] * w[c];
    h1[o] = s > 0.0f ? s : 0.0f;
  }
  __syncthreads();
  if (t < 256) {
    float s = b1[t];
    const float* w = W1 + (size_t)t * 512;
#pragma unroll 8
    for (int c = 0; c < 512; ++c) s += h1[c] * w[c];
    h2[t] = s > 0.0f ? s : 0.0f;
  }
  __syncthreads();
  if (t < 8) {
    float s = b2[t];
    const float* w = W2 + (size_t)t * 256;
#pragma unroll 8
    for (int c = 0; c < 256; ++c) s += h2[c] * w[c];
    out[g * 8 + t] = s;
  }
}

// ---------------------------------------------------------------------------
// Host-side launch
// ---------------------------------------------------------------------------
extern "C" void kernel_launch(void* const* d_in, const int* in_sizes, int n_in,
                              void* d_out, int out_size, void* d_ws, size_t ws_size,
                              hipStream_t stream) {
  (void)in_sizes; (void)n_in; (void)out_size; (void)ws_size;

  // ---- inputs (setup_inputs dict order, nested params flattened) ----
  const float* x      = (const float*)d_in[0];
  const float* pos    = (const float*)d_in[1];
  /* batch d_in[2] unused: block-contiguous equal-size graphs */
  const float* c1W0 = (const float*)d_in[3];  const float* c1b0 = (const float*)d_in[4];
  const float* c1g0 = (const float*)d_in[5];  const float* c1be0= (const float*)d_in[6];
  const float* c1mu0= (const float*)d_in[7];  const float* c1va0= (const float*)d_in[8];
  const float* c1W1 = (const float*)d_in[9];  const float* c1b1 = (const float*)d_in[10];
  const float* c1g1 = (const float*)d_in[11]; const float* c1be1= (const float*)d_in[12];
  const float* c1mu1= (const float*)d_in[13]; const float* c1va1= (const float*)d_in[14];
  const float* c1W2 = (const float*)d_in[15]; const float* c1b2 = (const float*)d_in[16];
  const float* c2W  = (const float*)d_in[17]; const float* c2b  = (const float*)d_in[18];
  const float* l1W  = (const float*)d_in[19]; const float* l1b  = (const float*)d_in[20];
  const float* mW0  = (const float*)d_in[21]; const float* mb0  = (const float*)d_in[22];
  const float* mg0  = (const float*)d_in[23]; const float* mbe0 = (const float*)d_in[24];
  const float* mmu0 = (const float*)d_in[25]; const float* mva0 = (const float*)d_in[26];
  const float* mW1  = (const float*)d_in[27]; const float* mb1  = (const float*)d_in[28];
  const float* mg1  = (const float*)d_in[29]; const float* mbe1 = (const float*)d_in[30];
  const float* mmu1 = (const float*)d_in[31]; const float* mva1 = (const float*)d_in[32];
  const float* mW2  = (const float*)d_in[33]; const float* mb2  = (const float*)d_in[34];

  // ---- workspace carve-out ----
  char* wp = (char*)d_ws;
  auto alloc = [&](size_t bytes) -> void* {
    void* p = (void*)wp;
    wp += (bytes + 255) & ~(size_t)255;
    return p;
  };
  _Float16* xxh   = (_Float16*)alloc((size_t)NTOT * 32 * 2);
  float*    d2a   = (float*)   alloc((size_t)NTOT * 4);
  int*      idx1  = (int*)     alloc((size_t)NTOT * KNN * 4);
  _Float16* feat  = (_Float16*)alloc((size_t)NTOT * 192 * 2);  // [x1 | x2] f16
  float*    x1f   = (float*)   alloc((size_t)NTOT * 64 * 4);
  float*    d2b   = (float*)   alloc((size_t)NTOT * 4);
  int*      idx2  = (int*)     alloc((size_t)NTOT * KNN * 4);
  float*    pool  = (float*)   alloc((size_t)B_GR * 1024 * 4);
  _Float16* fW0   = (_Float16*)alloc(64 * 32 * 2);   float* fb0 = (float*)alloc(64 * 4);
  _Float16* fW1   = (_Float16*)alloc(64 * 64 * 2);   float* fb1 = (float*)alloc(64 * 4);
  _Float16* fW2   = (_Float16*)alloc(64 * 64 * 2);   float* fb2 = (float*)alloc(64 * 4);
  _Float16* fWc2  = (_Float16*)alloc(128 * 128 * 2); float* fbc2= (float*)alloc(128 * 4);
  _Float16* fWl1  = (_Float16*)alloc(1024 * 192 * 2);float* fbl1= (float*)alloc(1024 * 4);
  float*    fWm0  = (float*)   alloc(512 * 1024 * 4);float* fbm0= (float*)alloc(512 * 4);
  float*    fWm1  = (float*)   alloc(256 * 512 * 4); float* fbm1= (float*)alloc(256 * 4);
  float*    fWm2  = (float*)   alloc(8 * 256 * 4);   float* fbm2= (float*)alloc(8 * 4);

  auto blk = [](int total) { return (total + 255) / 256; };

  // ---- fold BN + convert weights ----
  k_fold_f16<<<blk(64 * 32), 256, 0, stream>>>(c1W0, c1b0, c1g0, c1be0, c1mu0, c1va0,
                                               64, 8, 32, fW0, fb0);
  k_fold_f16<<<blk(64 * 64), 256, 0, stream>>>(c1W1, c1b1, c1g1, c1be1, c1mu1, c1va1,
                                               64, 64, 64, fW1, fb1);
  k_fold_f16<<<blk(64 * 64), 256, 0, stream>>>(c1W2, c1b2, nullptr, nullptr, nullptr, nullptr,
                                               64, 64, 64, fW2, fb2);
  k_fold_f16<<<blk(128 * 128), 256, 0, stream>>>(c2W, c2b, nullptr, nullptr, nullptr, nullptr,
                                                 128, 128, 128, fWc2, fbc2);
  k_fold_f16<<<blk(1024 * 192), 256, 0, stream>>>(l1W, l1b, nullptr, nullptr, nullptr, nullptr,
                                                  1024, 192, 192, fWl1, fbl1);
  k_fold_f32<<<blk(512 * 1024), 256, 0, stream>>>(mW0, mb0, mg0, mbe0, mmu0, mva0,
                                                  512, 1024, fWm0, fbm0);
  k_fold_f32<<<blk(256 * 512), 256, 0, stream>>>(mW1, mb1, mg1, mbe1, mmu1, mva1,
                                                 256, 512, fWm1, fbm1);
  k_fold_f32<<<blk(8 * 256), 256, 0, stream>>>(mW2, mb2, nullptr, nullptr, nullptr, nullptr,
                                               8, 256, fWm2, fbm2);

  // ---- pipeline ----
  k_pack_xx<<<256, 256, 0, stream>>>(x, pos, xxh, d2a);
  k_knn<1><<<B_GR * 64, 32, 0, stream>>>(xxh, 32, d2a, idx1);
  k_conv1<<<NTOT / 16, 32, 0, stream>>>(x, pos, idx1, fW0, fb0, fW1, fb1, fW2, fb2,
                                        x1f, feat);
  k_rownorm64<<<256, 256, 0, stream>>>(x1f, d2b);
  k_knn<2><<<B_GR * 64, 32, 0, stream>>>(feat, 192, d2b, idx2);
  k_conv2<<<NTOT / 16, 32, 0, stream>>>(x1f, idx2, fWc2, fbc2, feat);
  k_lin1max<<<B_GR * 64, 32, 0, stream>>>(feat, fWl1, fbl1, pool);
  k_mlp_head<<<B_GR, 256, 0, stream>>>(pool, fWm0, fbm0, fWm1, fbm1, fWm2, fbm2,
                                       (float*)d_out);
}
